// mHC_mlp_81346680586478
// MI455X (gfx1250) — matrix operand order repaired and checked by reading, not compiled, not run on hardware
//
#include <hip/hip_runtime.h>
#include <math.h>

// ---------------------------------------------------------------------------
// mHC MLP block for MI455X (gfx1250), wave32 + WMMA bf16x2-split GEMMs.
// Pipeline:
//   K1 rowstats : rms, H_pre/H_post/H_res (sinkhorn), x_pre        (mem-bound)
//   K2 gemm1    : G1 = gelu(x_pre @ W1 + b1)   [8192x2048 @ 2048x8192]
//   K3 gemm2    : Hbar = G1 @ W2 + b2          [8192x8192 @ 8192x2048]
//   K4 epilogue : out[b,n,:] = Hpost[b,n]*Hbar[b,:] + sum_m Hres[b,n,m]*x[b,m,:]
// GEMMs software-pipeline the next k-tile through registers so global-load
// latency overlaps the WMMA stream.
// ---------------------------------------------------------------------------

#define BATCH 8192
#define NN    4
#define DD    2048
#define DFF   8192
#define ND    8192   // N*D

typedef __attribute__((ext_vector_type(16))) __bf16        bf16x16;
typedef __attribute__((ext_vector_type(8)))  float         f32x8;
typedef __attribute__((ext_vector_type(4)))  unsigned int  u32x4;

__device__ __forceinline__ unsigned short f32_bf16_rne(float f) {
    unsigned int u = __float_as_uint(f);
    unsigned int r = 0x7FFFu + ((u >> 16) & 1u);
    return (unsigned short)((u + r) >> 16);
}
__device__ __forceinline__ float bf16_f32(unsigned short h) {
    return __uint_as_float(((unsigned int)h) << 16);
}

// ---------------------------------------------------------------------------
// Kernel 1: per-row statistics. 8 waves per block, one wave per batch row.
// ---------------------------------------------------------------------------
__global__ __launch_bounds__(256)
void rowstats_kernel(const float* __restrict__ x, const float* __restrict__ w,
                     const float* __restrict__ th_pre, const float* __restrict__ th_post,
                     const float* __restrict__ th_res,
                     const float* __restrict__ a_pre_p, const float* __restrict__ a_post_p,
                     const float* __restrict__ a_res_p,
                     const float* __restrict__ b_pre, const float* __restrict__ b_post,
                     const float* __restrict__ b_res,
                     float* __restrict__ x_pre, float* __restrict__ Hpost,
                     float* __restrict__ Hres)
{
    const int wave = threadIdx.x >> 5;
    const int lane = threadIdx.x & 31;
    const int b = blockIdx.x * 8 + wave;
    const float* xr = x + (size_t)b * ND;

    float ss = 0.0f;
    float s[24];
#pragma unroll
    for (int j = 0; j < 24; ++j) s[j] = 0.0f;

    for (int i = lane; i < ND; i += 32) {
        float xv = xr[i];
        ss += xv * xv;
        float xw = xv * w[i];
        const float4 tp = *(const float4*)(th_pre + (size_t)i * 4);
        s[0] += xw * tp.x; s[1] += xw * tp.y; s[2] += xw * tp.z; s[3] += xw * tp.w;
        const float4 tq = *(const float4*)(th_post + (size_t)i * 4);
        s[4] += xw * tq.x; s[5] += xw * tq.y; s[6] += xw * tq.z; s[7] += xw * tq.w;
        const float4* tr = (const float4*)(th_res + (size_t)i * 16);
#pragma unroll
        for (int q = 0; q < 4; ++q) {
            float4 t = tr[q];
            s[8 + q * 4 + 0] += xw * t.x;
            s[8 + q * 4 + 1] += xw * t.y;
            s[8 + q * 4 + 2] += xw * t.z;
            s[8 + q * 4 + 3] += xw * t.w;
        }
    }
    // wave32 butterfly reduce -> all lanes hold full sums
#pragma unroll
    for (int msk = 16; msk > 0; msk >>= 1) {
        ss += __shfl_xor(ss, msk);
#pragma unroll
        for (int j = 0; j < 24; ++j) s[j] += __shfl_xor(s[j], msk);
    }

    const float rms   = rsqrtf(ss * (1.0f / (float)ND) + 1.1920929e-7f);
    const float a_pre = a_pre_p[0], a_post = a_post_p[0], a_res = a_res_p[0];

    float hpre[4], hpost[4];
#pragma unroll
    for (int n = 0; n < 4; ++n) {
        hpre[n]  = 1.0f / (1.0f + expf(-(a_pre * rms * s[n] + b_pre[n])));
        hpost[n] = 2.0f / (1.0f + expf(-(a_post * rms * s[4 + n] + b_post[n])));
    }
    // Sinkhorn-Knopp on exp(alpha*rms*s_res + b_res), 20 iterations
    float m[16];
#pragma unroll
    for (int j = 0; j < 16; ++j) m[j] = expf(a_res * rms * s[8 + j] + b_res[j]);
#pragma unroll
    for (int t = 0; t < 20; ++t) {
#pragma unroll
        for (int r = 0; r < 4; ++r) {
            float inv = 1.0f / (m[r*4] + m[r*4+1] + m[r*4+2] + m[r*4+3]);
            m[r*4] *= inv; m[r*4+1] *= inv; m[r*4+2] *= inv; m[r*4+3] *= inv;
        }
#pragma unroll
        for (int c = 0; c < 4; ++c) {
            float inv = 1.0f / (m[c] + m[c+4] + m[c+8] + m[c+12]);
            m[c] *= inv; m[c+4] *= inv; m[c+8] *= inv; m[c+12] *= inv;
        }
    }
    if (lane == 0) {
#pragma unroll
        for (int n = 0; n < 4; ++n) Hpost[(size_t)b * 4 + n] = hpost[n];
#pragma unroll
        for (int j = 0; j < 16; ++j) Hres[(size_t)b * 16 + j] = m[j];
    }
    // x_pre = sum_n hpre[n] * x[b,n,:]
    for (int d = lane; d < DD; d += 32) {
        float v = hpre[0] * xr[d] + hpre[1] * xr[DD + d] +
                  hpre[2] * xr[2 * DD + d] + hpre[3] * xr[3 * DD + d];
        x_pre[(size_t)b * DD + d] = v;
    }
}

// ---------------------------------------------------------------------------
// Kernel 2/3: fp32 GEMM via bf16x2 split on v_wmma_f32_16x16x32_bf16.
// Block tile 128x128, BK=32, 8 waves (2x4), wave tile 64x32 (4x2 WMMA tiles).
// C = act(A @ B + bias), A[MxK] rm, B[KxN] rm, C[MxN] rm.
// MFAST: block index order (M fastest -> N-slab of B stays L2 resident).
// Next k-tile is register-staged so global-load latency hides under WMMAs.
// ---------------------------------------------------------------------------
template <bool GELU, bool MFAST>
__global__ __launch_bounds__(256)
void gemm_bf16x2(const float* __restrict__ A, const float* __restrict__ B,
                 const float* __restrict__ bias, float* __restrict__ C,
                 int M, int N, int K)
{
    __shared__ unsigned short sAh[128 * 40];
    __shared__ unsigned short sAl[128 * 40];
    __shared__ unsigned short sBh[128 * 40];
    __shared__ unsigned short sBl[128 * 40];

    const int tid = threadIdx.x;
    const int mblocks = M >> 7, nblocks = N >> 7;
    const int mb = MFAST ? ((int)blockIdx.x % mblocks) : ((int)blockIdx.x / nblocks);
    const int nb = MFAST ? ((int)blockIdx.x / mblocks) : ((int)blockIdx.x % nblocks);
    const int m0 = mb << 7, n0 = nb << 7;

    const int wave = tid >> 5, lane = tid & 31;
    const int wm = wave >> 2, wn = wave & 3;       // 2 x 4 wave grid
    const int lm = lane & 15, lh = lane >> 4;

    f32x8 acc[4][2];
#pragma unroll
    for (int i = 0; i < 4; ++i)
#pragma unroll
        for (int j = 0; j < 2; ++j) acc[i][j] = (f32x8)(0.0f);

    // fill mappings
    const int ar = tid >> 3, ac4 = (tid & 7) << 2;   // A: 4 rows per thread (stride 32)
    const int bn = tid & 127, bkh = tid >> 7;        // B: 16 k-values at fixed n

    // ---- preload tile k0=0 into registers ----
    float4 aReg[4];
    float  bReg[16];
#pragma unroll
    for (int i = 0; i < 4; ++i)
        aReg[i] = *(const float4*)(A + (size_t)(m0 + ar + 32 * i) * K + ac4);
#pragma unroll
    for (int i = 0; i < 16; ++i)
        bReg[i] = B[(size_t)(bkh * 16 + i) * N + n0 + bn];

    for (int k0 = 0; k0 < K; k0 += 32) {
        // ---- convert staged tile -> LDS (A row-major [m][k], B transposed [n][k]) ----
#pragma unroll
        for (int i = 0; i < 4; ++i) {
            const float4 v = aReg[i];
            unsigned short h0 = f32_bf16_rne(v.x), h1 = f32_bf16_rne(v.y);
            unsigned short h2 = f32_bf16_rne(v.z), h3 = f32_bf16_rne(v.w);
            unsigned short l0 = f32_bf16_rne(v.x - bf16_f32(h0));
            unsigned short l1 = f32_bf16_rne(v.y - bf16_f32(h1));
            unsigned short l2 = f32_bf16_rne(v.z - bf16_f32(h2));
            unsigned short l3 = f32_bf16_rne(v.w - bf16_f32(h3));
            uint2 ph; ph.x = (unsigned)h0 | ((unsigned)h1 << 16); ph.y = (unsigned)h2 | ((unsigned)h3 << 16);
            uint2 pl; pl.x = (unsigned)l0 | ((unsigned)l1 << 16); pl.y = (unsigned)l2 | ((unsigned)l3 << 16);
            *(uint2*)&sAh[(ar + 32 * i) * 40 + ac4] = ph;
            *(uint2*)&sAl[(ar + 32 * i) * 40 + ac4] = pl;
        }
        {
            unsigned int hw[8], lw[8];
#pragma unroll
            for (int i = 0; i < 16; ++i) {
                float v = bReg[i];
                unsigned short h = f32_bf16_rne(v);
                unsigned short l = f32_bf16_rne(v - bf16_f32(h));
                if (i & 1) { hw[i >> 1] |= ((unsigned)h) << 16; lw[i >> 1] |= ((unsigned)l) << 16; }
                else       { hw[i >> 1]  = h;                   lw[i >> 1]  = l; }
            }
            u32x4* dh = (u32x4*)&sBh[bn * 40 + bkh * 16];
            u32x4* dl = (u32x4*)&sBl[bn * 40 + bkh * 16];
            dh[0] = (u32x4){hw[0], hw[1], hw[2], hw[3]};
            dh[1] = (u32x4){hw[4], hw[5], hw[6], hw[7]};
            dl[0] = (u32x4){lw[0], lw[1], lw[2], lw[3]};
            dl[1] = (u32x4){lw[4], lw[5], lw[6], lw[7]};
        }
        __syncthreads();

        // ---- issue global loads for next k-tile; latency hides under WMMAs ----
        const int k1 = k0 + 32;
        if (k1 < K) {
#pragma unroll
            for (int i = 0; i < 4; ++i)
                aReg[i] = *(const float4*)(A + (size_t)(m0 + ar + 32 * i) * K + k1 + ac4);
#pragma unroll
            for (int i = 0; i < 16; ++i)
                bReg[i] = B[(size_t)(k1 + bkh * 16 + i) * N + n0 + bn];
        }

        // ---- fragments (ISA 16-bit A/B layouts, 16B-aligned ds_load_b128) ----
        bf16x16 afh[4], afl[4], bfh[2], bfl[2];
#pragma unroll
        for (int t = 0; t < 4; ++t) {
            int base = (wm * 64 + t * 16 + lm) * 40 + 8 * lh;
            union { u32x4 q[2]; bf16x16 v; } fh, fl;
            fh.q[0] = *(const u32x4*)&sAh[base];
            fh.q[1] = *(const u32x4*)&sAh[base + 16];
            fl.q[0] = *(const u32x4*)&sAl[base];
            fl.q[1] = *(const u32x4*)&sAl[base + 16];
            afh[t] = fh.v; afl[t] = fl.v;
        }
#pragma unroll
        for (int t = 0; t < 2; ++t) {
            int base = (wn * 32 + t * 16 + lm) * 40 + 8 * lh;
            union { u32x4 q[2]; bf16x16 v; } fh, fl;
            fh.q[0] = *(const u32x4*)&sBh[base];
            fh.q[1] = *(const u32x4*)&sBh[base + 16];
            fl.q[0] = *(const u32x4*)&sBl[base];
            fl.q[1] = *(const u32x4*)&sBl[base + 16];
            bfh[t] = fh.v; bfl[t] = fl.v;
        }
        // ---- 24 WMMAs: hi*hi + hi*lo + lo*hi (~fp32 accuracy) ----
#pragma unroll
        for (int tm = 0; tm < 4; ++tm)
#pragma unroll
            for (int tn = 0; tn < 2; ++tn) {
                acc[tm][tn] = __builtin_amdgcn_wmma_f32_16x16x32_bf16(
                    false, afh[tm], false, bfh[tn], (short)0, acc[tm][tn], false, false);
                acc[tm][tn] = __builtin_amdgcn_wmma_f32_16x16x32_bf16(
                    false, afh[tm], false, bfl[tn], (short)0, acc[tm][tn], false, false);
                acc[tm][tn] = __builtin_amdgcn_wmma_f32_16x16x32_bf16(
                    false, afl[tm], false, bfh[tn], (short)0, acc[tm][tn], false, false);
            }
        __syncthreads();
    }

    // ---- epilogue: bias (+ exact GELU), store ----
#pragma unroll
    for (int tm = 0; tm < 4; ++tm)
#pragma unroll
        for (int tn = 0; tn < 2; ++tn) {
            int n = n0 + wn * 32 + tn * 16 + lm;
            float bv = bias[n];
#pragma unroll
            for (int r = 0; r < 8; ++r) {
                int mrow = m0 + wm * 64 + tm * 16 + 8 * lh + r;
                float v = acc[tm][tn][r] + bv;
                if (GELU) v = 0.5f * v * (1.0f + erff(v * 0.70710678118654752f));
                C[(size_t)mrow * N + n] = v;
            }
        }
}

// ---------------------------------------------------------------------------
// Kernel 4: out[b,n,d] = Hpost[b,n]*Hbar[b,d] + sum_m Hres[b,n,m]*x[b,m,d]
// ---------------------------------------------------------------------------
__global__ __launch_bounds__(256)
void epilogue_kernel(const float* __restrict__ x, const float* __restrict__ Hbar,
                     const float* __restrict__ Hpost, const float* __restrict__ Hres,
                     float* __restrict__ out)
{
    int idx = blockIdx.x * 256 + threadIdx.x;   // over B * (D/4)
    int b = idx >> 9;
    int d = (idx & 511) << 2;
    const float4 hb = *(const float4*)(Hbar + (size_t)b * DD + d);
    float4 xm[4];
#pragma unroll
    for (int m = 0; m < 4; ++m)
        xm[m] = *(const float4*)(x + ((size_t)b * 4 + m) * DD + d);
    float hp[4], hr[16];
#pragma unroll
    for (int n = 0; n < 4; ++n) hp[n] = Hpost[(size_t)b * 4 + n];
#pragma unroll
    for (int j = 0; j < 16; ++j) hr[j] = Hres[(size_t)b * 16 + j];
#pragma unroll
    for (int n = 0; n < 4; ++n) {
        float4 o;
        o.x = hp[n] * hb.x; o.y = hp[n] * hb.y; o.z = hp[n] * hb.z; o.w = hp[n] * hb.w;
#pragma unroll
        for (int m = 0; m < 4; ++m) {
            float c = hr[n * 4 + m];
            o.x += c * xm[m].x; o.y += c * xm[m].y; o.z += c * xm[m].z; o.w += c * xm[m].w;
        }
        *(float4*)(out + ((size_t)b * 4 + n) * DD + d) = o;
    }
}

// ---------------------------------------------------------------------------
extern "C" void kernel_launch(void* const* d_in, const int* in_sizes, int n_in,
                              void* d_out, int out_size, void* d_ws, size_t ws_size,
                              hipStream_t stream)
{
    const float* x       = (const float*)d_in[0];
    const float* rms_w   = (const float*)d_in[1];
    const float* th_pre  = (const float*)d_in[2];
    const float* th_post = (const float*)d_in[3];
    const float* th_res  = (const float*)d_in[4];
    const float* a_pre   = (const float*)d_in[5];
    const float* a_post  = (const float*)d_in[6];
    const float* a_res   = (const float*)d_in[7];
    const float* b_pre   = (const float*)d_in[8];
    const float* b_post  = (const float*)d_in[9];
    const float* b_res   = (const float*)d_in[10];
    const float* W1      = (const float*)d_in[11];
    const float* b1      = (const float*)d_in[12];
    const float* W2      = (const float*)d_in[13];
    const float* b2      = (const float*)d_in[14];

    float* ws    = (float*)d_ws;
    float* x_pre = ws;                                   // B*D      = 16,777,216 f
    float* G1    = x_pre + (size_t)BATCH * DD;           // B*DFF    = 67,108,864 f
    float* Hbar  = G1 + (size_t)BATCH * DFF;             // B*D
    float* Hpost = Hbar + (size_t)BATCH * DD;            // B*4
    float* Hres  = Hpost + (size_t)BATCH * 4;            // B*16

    // K1: 8 rows per block
    rowstats_kernel<<<BATCH / 8, 256, 0, stream>>>(
        x, rms_w, th_pre, th_post, th_res, a_pre, a_post, a_res,
        b_pre, b_post, b_res, x_pre, Hpost, Hres);

    // K2: G1 = gelu(x_pre @ W1 + b1)   M=8192 N=8192 K=2048, M-fastest order
    gemm_bf16x2<true, true><<<(BATCH / 128) * (DFF / 128), 256, 0, stream>>>(
        x_pre, W1, b1, G1, BATCH, DFF, DD);

    // K3: Hbar = G1 @ W2 + b2          M=8192 N=2048 K=8192, N-fastest order
    gemm_bf16x2<false, false><<<(BATCH / 128) * (DD / 128), 256, 0, stream>>>(
        G1, W2, b2, Hbar, BATCH, DD, DFF);

    // K4: combine
    epilogue_kernel<<<(BATCH * (DD / 4)) / 256, 256, 0, stream>>>(
        x, Hbar, Hpost, Hres, (float*)d_out);
}
